// IntrospectiveAlignmentLayer_11158325035251
// MI455X (gfx1250) — compile-verified, matrix operand order checked by reading, not run on hardware
//
#include <hip/hip_runtime.h>
#include <hip/hip_bf16.h>

typedef __attribute__((ext_vector_type(16))) _Float16 v16h;
typedef __attribute__((ext_vector_type(8)))  _Float16 v8h;   // 16 bytes
typedef __attribute__((ext_vector_type(8)))  float    v8f;

#define DEV __device__ __forceinline__

// ---------------------------------------------------------------------------
// Fragment loader: 16x32 f16 fragment (A: rows=M, B: rows=N), K-major rows.
// ISA layout (CDNA5 ISA): lane L -> row = L&15, half = L>>4;
// VGPR0..3 hold K = half*8 + {0..7}; VGPR4..7 hold K = 16 + half*8 + {0..7}.
// Both chunks are contiguous 16B -> two global_load_b128 per lane.
// ---------------------------------------------------------------------------
DEV v16h frag_ld(const _Float16* base, int ld, int lane) {
    int r = lane & 15, h = (lane >> 4) & 1;
    const _Float16* p = base + (long)r * ld + h * 8;
    union { v16h v; v8h x[2]; } u;
    u.x[0] = *(const v8h*)(p);
    u.x[1] = *(const v8h*)(p + 16);
    return u.v;
}

DEV v8f wmma_f16(v16h a, v16h b, v8f c) {
    return __builtin_amdgcn_wmma_f32_16x16x32_f16(false, a, false, b,
                                                  (short)0, c, false, false);
}

DEV float sigmoidf_(float x) { return 1.0f / (1.0f + __expf(-x)); }

// ---------------------------------------------------------------------------
// Generic NT WMMA GEMM: C = act(A[M,K] * B[N,K]^T + bias + bias2)
// A,B f16; C f32 (optional); C2 f16 (optional, own ldc2 for strided concat).
// Block: 256 threads = 8 waves (2 M x 4 N), wave tile 32x32, block tile 64x128.
// All problem sizes here are exact multiples (M%64==0, N%128==0, K%32==0).
// ---------------------------------------------------------------------------
__global__ void k_gemm_nt(const _Float16* __restrict__ A,
                          const _Float16* __restrict__ B,
                          float* __restrict__ C, _Float16* __restrict__ C2,
                          const float* __restrict__ bias,
                          const float* __restrict__ bias2,
                          int M, int N, int K,
                          int lda, int ldb, int ldc, int ldc2,
                          long sA, long sB, long sC, long sC2, int act)
{
    const int lane = threadIdx.x & 31;
    const int wave = threadIdx.x >> 5;
    const int wm = wave >> 2, wn = wave & 3;
    const long bz = blockIdx.z;
    const _Float16* Ab = A + bz * sA;
    const _Float16* Bb = B + bz * sB;
    const int m0 = blockIdx.y * 64 + wm * 32;
    const int n0 = blockIdx.x * 128 + wn * 32;
    if (m0 >= M || n0 >= N) return;   // wave-uniform (never taken for our sizes)

    v8f acc[2][2] = {};
    const _Float16* ap0 = Ab + (long)m0 * lda;
    const _Float16* ap1 = Ab + (long)(m0 + 16) * lda;
    const _Float16* bp0 = Bb + (long)n0 * ldb;
    const _Float16* bp1 = Bb + (long)(n0 + 16) * ldb;
    for (int k = 0; k < K; k += 32) {
        __builtin_prefetch(bp0 + k + 64, 0, 1);   // global_prefetch_b8
        v16h a0 = frag_ld(ap0 + k, lda, lane);
        v16h a1 = frag_ld(ap1 + k, lda, lane);
        v16h b0 = frag_ld(bp0 + k, ldb, lane);
        v16h b1 = frag_ld(bp1 + k, ldb, lane);
        acc[0][0] = wmma_f16(a0, b0, acc[0][0]);
        acc[0][1] = wmma_f16(a0, b1, acc[0][1]);
        acc[1][0] = wmma_f16(a1, b0, acc[1][0]);
        acc[1][1] = wmma_f16(a1, b1, acc[1][1]);
    }

    const int col = lane & 15, half = lane >> 4;
#pragma unroll
    for (int mi = 0; mi < 2; ++mi) {
#pragma unroll
        for (int ni = 0; ni < 2; ++ni) {
            int n = n0 + ni * 16 + col;
            float bb = 0.0f;
            if (bias)  bb += bias[n];
            if (bias2) bb += bias2[n];
#pragma unroll
            for (int r = 0; r < 8; ++r) {
                int m = m0 + mi * 16 + r + 8 * half;
                float v = acc[mi][ni][r] + bb;
                if (act) v = tanhf(v);
                if (C)  C [bz * sC  + (long)m * ldc  + n] = v;
                if (C2) C2[bz * sC2 + (long)m * ldc2 + n] = (_Float16)v;
            }
        }
    }
}

// ---------------------------------------------------------------------------
// Row softmax over N columns, optional band mask (value zeroed, still
// participates — matches reference S = S*mask then softmax). Writes f32
// in place and an optional f16 copy.
// ---------------------------------------------------------------------------
DEV float wred_max(float v) { for (int o = 16; o > 0; o >>= 1) v = fmaxf(v, __shfl_xor(v, o, 32)); return v; }
DEV float wred_sum(float v) { for (int o = 16; o > 0; o >>= 1) v += __shfl_xor(v, o, 32); return v; }

__global__ void k_softmax_rows(float* __restrict__ S, _Float16* __restrict__ P16,
                               int N, int rowsPerBatch, const int* __restrict__ band_ptr)
{
    const long row = blockIdx.x;
    float* p = S + row * (long)N;
    const int i = (int)(row % rowsPerBatch);
    const int band = band_ptr ? *band_ptr : -1;
    const int lane = threadIdx.x & 31, wave = threadIdx.x >> 5;
    __shared__ float sred[8];
    __shared__ float sval;

    float mx = -3.0e38f;
    for (int j = threadIdx.x; j < N; j += blockDim.x) {
        float v = p[j];
        if (band >= 0) { int d = i - j; if (d < 0) d = -d; if (d > band) v = 0.0f; }
        mx = fmaxf(mx, v);
    }
    mx = wred_max(mx);
    if (lane == 0) sred[wave] = mx;
    __syncthreads();
    if (threadIdx.x == 0) { float m = sred[0]; for (int w = 1; w < 8; ++w) m = fmaxf(m, sred[w]); sval = m; }
    __syncthreads();
    mx = sval;
    __syncthreads();

    float sum = 0.0f;
    for (int j = threadIdx.x; j < N; j += blockDim.x) {
        float v = p[j];
        if (band >= 0) { int d = i - j; if (d < 0) d = -d; if (d > band) v = 0.0f; }
        sum += __expf(v - mx);
    }
    sum = wred_sum(sum);
    if (lane == 0) sred[wave] = sum;
    __syncthreads();
    if (threadIdx.x == 0) { float m = 0.0f; for (int w = 0; w < 8; ++w) m += sred[w]; sval = m; }
    __syncthreads();
    const float inv = 1.0f / sval;

    for (int j = threadIdx.x; j < N; j += blockDim.x) {
        float v = p[j];
        if (band >= 0) { int d = i - j; if (d < 0) d = -d; if (d > band) v = 0.0f; }
        float o = __expf(v - mx) * inv;
        p[j] = o;
        if (P16) P16[row * (long)N + j] = (_Float16)o;
    }
}

// ---------------------------------------------------------------------------
// Batched f16 transpose: src [z,R,C] -> dst [z,C,R], 32x32 LDS tiles.
// ---------------------------------------------------------------------------
__global__ void k_transpose_h(const _Float16* __restrict__ src, _Float16* __restrict__ dst,
                              int R, int C)
{
    __shared__ _Float16 tile[32][33];
    const long zoff = (long)blockIdx.z * R * C;
    const _Float16* s = src + zoff;
    _Float16* d = dst + zoff;
    int c0 = blockIdx.x * 32, r0 = blockIdx.y * 32;
#pragma unroll
    for (int k = 0; k < 32; k += 8)
        tile[threadIdx.y + k][threadIdx.x] = s[(long)(r0 + threadIdx.y + k) * C + c0 + threadIdx.x];
    __syncthreads();
#pragma unroll
    for (int k = 0; k < 32; k += 8)
        d[(long)(c0 + threadIdx.y + k) * R + r0 + threadIdx.x] = tile[threadIdx.x][threadIdx.y + k];
}

// f32 -> f16 grid-stride convert
__global__ void k_cvt(const float* __restrict__ src, _Float16* __restrict__ dst, long n)
{
    long i = (long)blockIdx.x * blockDim.x + threadIdx.x;
    long st = (long)gridDim.x * blockDim.x;
    for (; i < n; i += st) dst[i] = (_Float16)src[i];
}

// tmp = concat(A, Hc1, A-Hc1, A*Hc1) over last dim; writes f32 + f16
__global__ void k_build_tmp(const float* __restrict__ A, const float* __restrict__ Hc1,
                            float* __restrict__ tmp, _Float16* __restrict__ tmph, long n)
{
    long i = (long)blockIdx.x * blockDim.x + threadIdx.x;
    long st = (long)gridDim.x * blockDim.x;
    for (; i < n; i += st) {
        long row = i >> 8;        // / 256
        int  d   = (int)(i & 255);
        float a = A[i], h = Hc1[i];
        long b = row * 1024;
        float v0 = a, v1 = h, v2 = a - h, v3 = a * h;
        tmp[b + d]        = v0;  tmph[b + d]        = (_Float16)v0;
        tmp[b + 256 + d]  = v1;  tmph[b + 256 + d]  = (_Float16)v1;
        tmp[b + 512 + d]  = v2;  tmph[b + 512 + d]  = (_Float16)v2;
        tmp[b + 768 + d]  = v3;  tmph[b + 768 + d]  = (_Float16)v3;
    }
}

// Yh[:, 1024:2048] = tmph  (B part is written by the GEMM C2 path)
__global__ void k_copy_tail(const _Float16* __restrict__ tmph, _Float16* __restrict__ Yh, long n)
{
    long i = (long)blockIdx.x * blockDim.x + threadIdx.x;
    long st = (long)gridDim.x * blockDim.x;
    for (; i < n; i += st) {
        long row = i >> 10;       // / 1024
        int  d   = (int)(i & 1023);
        Yh[row * 2048 + 1024 + d] = tmph[i];
    }
}

// ---------------------------------------------------------------------------
// Persistent bidirectional LSTM step kernel. grid.x = 2 (direction),
// 1024 threads = 32 waves. Per step: g = h @ Whh^T via WMMA
// (M=16 padded batch, N=1024 gates, K=256), gate exchange via LDS,
// h/c state LDS-resident. Whh is f16 and L2-resident (512 KB).
// ---------------------------------------------------------------------------
__global__ void __launch_bounds__(1024)
k_lstm_dir(const float* __restrict__ xg0, const float* __restrict__ xg1,
           const _Float16* __restrict__ whh2,   // [2,1024,256] f16
           float* __restrict__ out_f32,         // [8,T,512] or null
           _Float16* __restrict__ out_f16,      // [8,T,512] or null
           int T)
{
    const int dir = blockIdx.x;
    const float* xg = dir ? xg1 : xg0;
    const _Float16* whh = whh2 + (long)dir * 1024 * 256;

    __shared__ _Float16 h_buf[16 * 256];   // rows 8..15 stay zero (M pad)
    __shared__ float    c_buf[8 * 256];
    __shared__ float    g_buf[8 * 1024];

    const int tid = threadIdx.x;
    for (int i = tid; i < 16 * 256; i += 1024) h_buf[i] = (_Float16)0.0f;
    for (int i = tid; i < 8 * 256;  i += 1024) c_buf[i] = 0.0f;
    __syncthreads();

    const int lane = tid & 31;
    const int wave = tid >> 5;             // 0..31
    const int n0 = wave * 32;              // two 16-wide N tiles per wave
    const int col = lane & 15, half = lane >> 4;

    for (int step = 0; step < T; ++step) {
        const int t = dir ? (T - 1 - step) : step;

        v8f acc0 = {}, acc1 = {};
        const _Float16* wp0 = whh + (long)n0 * 256;
        const _Float16* wp1 = whh + (long)(n0 + 16) * 256;
#pragma unroll
        for (int k = 0; k < 256; k += 32) {
            v16h a  = frag_ld(h_buf + k, 256, lane);   // flat->LDS loads
            v16h b0 = frag_ld(wp0 + k, 256, lane);
            v16h b1 = frag_ld(wp1 + k, 256, lane);
            acc0 = wmma_f16(a, b0, acc0);
            acc1 = wmma_f16(a, b1, acc1);
        }
#pragma unroll
        for (int r = 0; r < 8; ++r) {
            int m = r + 8 * half;
            if (m < 8) {
                g_buf[m * 1024 + n0 + col]      = acc0[r];
                g_buf[m * 1024 + n0 + 16 + col] = acc1[r];
            }
        }
        __syncthreads();

        // gate nonlinearity + state update: 8x256 units over 1024 threads
        for (int p = tid; p < 8 * 256; p += 1024) {
            int b = p >> 8, u = p & 255;
            const float* xr = xg + ((long)b * T + t) * 1024;
            float gi = g_buf[b * 1024 + u]        + xr[u];
            float gf = g_buf[b * 1024 + 256 + u]  + xr[256 + u];
            float gg = g_buf[b * 1024 + 512 + u]  + xr[512 + u];
            float go = g_buf[b * 1024 + 768 + u]  + xr[768 + u];
            float c = sigmoidf_(gf) * c_buf[p] + sigmoidf_(gi) * tanhf(gg);
            float h = sigmoidf_(go) * tanhf(c);
            c_buf[p] = c;
            h_buf[b * 256 + u] = (_Float16)h;
            long o = ((long)b * T + t) * 512 + dir * 256 + u;
            if (out_f32) out_f32[o] = h;
            if (out_f16) out_f16[o] = (_Float16)h;
        }
        __syncthreads();
    }
}

// ---------------------------------------------------------------------------
// Host orchestration
// ---------------------------------------------------------------------------
static void gemm_nt(hipStream_t s, const _Float16* A, const _Float16* B,
                    float* C, _Float16* C2, const float* bias, const float* bias2,
                    int M, int N, int K, int lda, int ldb, int ldc, int ldc2,
                    long sA, long sB, long sC, long sC2, int batch, int act)
{
    dim3 g(N / 128, M / 64, batch), b(256);
    k_gemm_nt<<<g, b, 0, s>>>(A, B, C, C2, bias, bias2, M, N, K,
                              lda, ldb, ldc, ldc2, sA, sB, sC, sC2, act);
}

extern "C" void kernel_launch(void* const* d_in, const int* in_sizes, int n_in,
                              void* d_out, int out_size, void* d_ws, size_t ws_size,
                              hipStream_t stream)
{
    (void)in_sizes; (void)n_in; (void)out_size; (void)ws_size;
    const float* Hq   = (const float*)d_in[0];
    const float* Hc   = (const float*)d_in[1];
    const float* W1   = (const float*)d_in[2];
    const float* b1   = (const float*)d_in[3];
    const float* W2   = (const float*)d_in[4];
    const float* b2   = (const float*)d_in[5];
    const float* Wih0 = (const float*)d_in[6];
    const float* Whh0 = (const float*)d_in[7];
    const float* bih0 = (const float*)d_in[8];
    const float* bhh0 = (const float*)d_in[9];
    const float* Wih  = (const float*)d_in[10];
    const float* Whh  = (const float*)d_in[11];
    const float* bih  = (const float*)d_in[12];
    const float* bhh  = (const float*)d_in[13];
    const int*   band = (const int*)d_in[14];
    float* out = (float*)d_out;

    const long B = 8, L = 1024, D = 256, H4 = 1024, D2 = 512, D8 = 2048, T = 1024;
    const long BL = B * L;                    // 8192
    const long LD = L * D;                    // 262144  per-batch [1024,256]
    const long LL = L * L;                    // 1048576 per-batch [1024,1024]

    char* ws = (char*)d_ws; size_t cur = 0;
    auto alloc = [&](size_t bytes) -> char* {
        char* p = ws + cur; cur += (bytes + 255) & ~(size_t)255; return p;
    };
    _Float16* Hqh    = (_Float16*)alloc(BL * D * 2);
    _Float16* Hch    = (_Float16*)alloc(BL * D * 2);
    _Float16* W1h    = (_Float16*)alloc(D * D * 2);
    _Float16* W2h    = (_Float16*)alloc(H4 * H4 * 2);
    _Float16* Wih0h  = (_Float16*)alloc(2 * H4 * D8 * 2);
    _Float16* Wihh   = (_Float16*)alloc(4 * 2 * H4 * D2 * 2);
    _Float16* WhhAll = (_Float16*)alloc(10 * H4 * D * 2);   // 5 layers x 2 dirs
    float*    Hc1    = (float*)   alloc(BL * D * 4);
    _Float16* Hq1h   = (_Float16*)alloc(BL * D * 2);
    _Float16* Hc1h   = (_Float16*)alloc(BL * D * 2);
    _Float16* Hq1Th  = (_Float16*)alloc(BL * D * 2);        // [b,256,1024]
    float*    Ebuf   = (float*)   alloc(B * LL * 4);        // E, reused as S
    _Float16* Ph     = (_Float16*)alloc(B * LL * 2);
    float*    Amat   = (float*)   alloc(BL * D * 4);
    float*    tmp    = (float*)   alloc(B * LL * 4);
    _Float16* tmph   = (_Float16*)alloc(B * LL * 2);
    _Float16* tmpTh  = (_Float16*)alloc(B * LL * 2);
    _Float16* P2h    = (_Float16*)alloc(B * LL * 2);
    _Float16* Yh     = (_Float16*)alloc(BL * D8 * 2);       // [b,t,2048]
    float*    xgd0   = (float*)   alloc(BL * H4 * 4);
    float*    xgd1   = (float*)   alloc(BL * H4 * 4);
    _Float16* layh0  = (_Float16*)alloc(BL * D2 * 2);
    _Float16* layh1  = (_Float16*)alloc(BL * D2 * 2);

    const dim3 cb(256); const dim3 cg(4096);
    // -------- f16 conversions of inputs / weights --------
    k_cvt<<<cg, cb, 0, stream>>>(Hq,   Hqh,   BL * D);
    k_cvt<<<cg, cb, 0, stream>>>(Hc,   Hch,   BL * D);
    k_cvt<<<cg, cb, 0, stream>>>(W1,   W1h,   D * D);
    k_cvt<<<cg, cb, 0, stream>>>(W2,   W2h,   H4 * H4);
    k_cvt<<<cg, cb, 0, stream>>>(Wih0, Wih0h, 2 * H4 * D8);
    k_cvt<<<cg, cb, 0, stream>>>(Wih,  Wihh,  4 * 2 * H4 * D2);
    k_cvt<<<cg, cb, 0, stream>>>(Whh0, WhhAll,              2 * H4 * D);
    k_cvt<<<cg, cb, 0, stream>>>(Whh,  WhhAll + 2 * H4 * D, 8 * H4 * D);

    // -------- Hq1 = tanh(Hq W1^T + b1), Hc1 likewise --------
    gemm_nt(stream, Hqh, W1h, nullptr, Hq1h, b1, nullptr,
            (int)BL, (int)D, (int)D, (int)D, (int)D, 0, (int)D, 0, 0, 0, 0, 1, 1);
    gemm_nt(stream, Hch, W1h, Hc1, Hc1h, b1, nullptr,
            (int)BL, (int)D, (int)D, (int)D, (int)D, (int)D, (int)D, 0, 0, 0, 0, 1, 1);

    // -------- E = Hc1 Hq1^T (batched NT), softmax rows --------
    gemm_nt(stream, Hc1h, Hq1h, Ebuf, nullptr, nullptr, nullptr,
            (int)L, (int)L, (int)D, (int)D, (int)D, (int)L, 0, LD, LD, LL, 0, 8, 0);
    k_softmax_rows<<<dim3((unsigned)(B * L)), dim3(256), 0, stream>>>(
        Ebuf, Ph, (int)L, (int)L, nullptr);

    // -------- A = P Hq1  via transposed B operand (NT again) --------
    k_transpose_h<<<dim3(8, 32, 8), dim3(32, 8), 0, stream>>>(Hq1h, Hq1Th, (int)L, (int)D);
    gemm_nt(stream, Ph, Hq1Th, Amat, nullptr, nullptr, nullptr,
            (int)L, (int)D, (int)L, (int)L, (int)L, (int)D, 0, LL, LD, LD, 0, 8, 0);

    // -------- tmp = concat(A, Hc1, A-Hc1, A*Hc1) --------
    k_build_tmp<<<cg, cb, 0, stream>>>(Amat, Hc1, tmp, tmph, BL * D);

    // -------- G = tanh(tmp W2^T + b2), f16 only --------
    gemm_nt(stream, tmph, W2h, nullptr, Ph /*reuse as Gh*/, b2, nullptr,
            (int)BL, (int)H4, (int)H4, (int)H4, (int)H4, 0, (int)H4, 0, 0, 0, 0, 1, 1);
    _Float16* Gh = Ph;   // Ph no longer needed after A-GEMM

    // -------- S = G G^T (native NT), banded softmax (mask multiplies by 0) --------
    gemm_nt(stream, Gh, Gh, Ebuf /*reuse as S*/, nullptr, nullptr, nullptr,
            (int)L, (int)L, (int)H4, (int)H4, (int)H4, (int)L, 0, LL, LL, LL, 0, 8, 0);
    k_softmax_rows<<<dim3((unsigned)(B * L)), dim3(256), 0, stream>>>(
        Ebuf, P2h, (int)L, (int)L, band);

    // -------- B = softmax(S) tmp  ->  written straight into Y[:, :1024] --------
    k_transpose_h<<<dim3(32, 32, 8), dim3(32, 8), 0, stream>>>(tmph, tmpTh, (int)L, (int)H4);
    gemm_nt(stream, P2h, tmpTh, nullptr, Yh, nullptr, nullptr,
            (int)L, (int)H4, (int)L, (int)L, (int)L, 0, (int)D8, LL, LL, 0, L * D8, 8, 0);
    k_copy_tail<<<cg, cb, 0, stream>>>(tmph, Yh, B * LL);

    // -------- 5-layer bidirectional LSTM --------
    const _Float16* cur_in = Yh;
    int curK = (int)D8;
    for (int l = 0; l < 5; ++l) {
        const _Float16* wih = (l == 0) ? Wih0h : (Wihh + (long)(l - 1) * 2 * H4 * D2);
        const float* bi = (l == 0) ? bih0 : (bih + (long)(l - 1) * 2 * H4);
        const float* bh = (l == 0) ? bhh0 : (bhh + (long)(l - 1) * 2 * H4);
        // xg per direction: [8192, 1024] = x @ Wih_d^T + (bih_d + bhh_d)
        for (int d = 0; d < 2; ++d) {
            float* xg = d ? xgd1 : xgd0;
            gemm_nt(stream, cur_in, wih + (long)d * H4 * curK,
                    xg, nullptr, bi + (long)d * H4, bh + (long)d * H4,
                    (int)BL, (int)H4, curK, curK, curK, (int)H4, 0,
                    0, 0, 0, 0, 1, 0);
        }
        const _Float16* whh_l = WhhAll + (long)l * 2 * H4 * D;
        bool last = (l == 4);
        _Float16* oh = last ? nullptr : ((l & 1) ? layh1 : layh0);
        float*    of = last ? out : nullptr;
        k_lstm_dir<<<dim3(2), dim3(1024), 0, stream>>>(xgd0, xgd1, whh_l, of, oh, (int)T);
        cur_in = oh;
        curK = (int)D2;
    }
}
